// DynamicMoERoutingLayer_58720792871363
// MI455X (gfx1250) — matrix operand order repaired and checked by reading, not compiled, not run on hardware
//
#include <hip/hip_runtime.h>
#include <hip/hip_bf16.h>
#include <math.h>

// ---------------- problem constants ----------------
#define BN   32      // batch
#define CIN  32
#define HW   128     // H = W = 128
#define COUT 64
#define EE   10      // experts
#define RDIM 128
#define HID  128
#define RD   64      // routing output dim
#define HO   63
#define WO   63
#define NPIX (HO*WO)             // 3969
#define KDIM (CIN*9)             // 288
#define NT64 ((NPIX + 63) / 64)  // 63 pixel tiles of 64

typedef __attribute__((ext_vector_type(16))) __bf16 v16bf;
typedef __attribute__((ext_vector_type(8)))  float  v8f;

__device__ __forceinline__ __bf16 f2bf(float f) {
    // round-to-nearest-even fp32 -> bf16
    union { float f; unsigned u; } cv; cv.f = f;
    unsigned u = cv.u;
    u += 0x7FFFu + ((u >> 16) & 1u);
    unsigned short hs = (unsigned short)(u >> 16);
    __bf16 out;
    __builtin_memcpy(&out, &hs, 2);
    return out;
}

// ---------------------------------------------------------------------------
// Kernel 1: routing MLP -> cosine sim -> softmax weights (one block per batch)
// ---------------------------------------------------------------------------
__global__ void routing_kernel(const float* __restrict__ rv,
                               const float* __restrict__ W1, const float* __restrict__ b1,
                               const float* __restrict__ W2, const float* __restrict__ b2,
                               const float* __restrict__ emb,
                               float* __restrict__ weights, float* __restrict__ wsum) {
    __shared__ float h[HID];
    __shared__ float r[RD];
    __shared__ float red[HID];
    __shared__ float sim[EE];
    const int b = blockIdx.x, tid = threadIdx.x;
    const float* rvb = rv + b * RDIM;

    // h = relu(rv @ W1.T + b1)
    float acc = b1[tid];
    for (int k = 0; k < RDIM; ++k) acc += rvb[k] * W1[tid * RDIM + k];
    h[tid] = fmaxf(acc, 0.0f);
    __syncthreads();

    // r = h @ W2.T + b2
    if (tid < RD) {
        float a2 = b2[tid];
        for (int k = 0; k < HID; ++k) a2 += h[k] * W2[tid * HID + k];
        r[tid] = a2;
    }
    __syncthreads();

    // ||r||
    red[tid] = (tid < RD) ? r[tid] * r[tid] : 0.0f;
    __syncthreads();
    for (int s = 64; s > 0; s >>= 1) {
        if (tid < s) red[tid] += red[tid + s];
        __syncthreads();
    }
    const float nrm = sqrtf(red[0]);

    // cosine similarity against each (row-normalized) embedding
    if (tid < EE) {
        float es = 0.f, dp = 0.f;
        for (int j = 0; j < RD; ++j) {
            float e = emb[tid * RD + j];
            es += e * e;
            dp += r[j] * e;
        }
        sim[tid] = dp / (nrm * sqrtf(es));
    }
    __syncthreads();

    // softmax over E=10 (tiny; single thread) + row-sum for the final divide
    if (tid == 0) {
        float mx = sim[0];
        for (int e = 1; e < EE; ++e) mx = fmaxf(mx, sim[e]);
        float ex[EE], s = 0.f;
        for (int e = 0; e < EE; ++e) { ex[e] = expf(sim[e] - mx); s += ex[e]; }
        float tot = 0.f;
        for (int e = 0; e < EE; ++e) {
            float w = ex[e] / s;
            weights[b * EE + e] = w;
            tot += w;
        }
        wsum[b] = tot;
    }
}

// ---------------------------------------------------------------------------
// Kernel 2: fold expert-weighted sum into per-batch filters/bias, and pack
// the mixed filters directly in bf16 WMMA A-fragment layout with a REORDERED
// reduction index k' = (kh*3+kw)*32 + ci, so every 32-wide K chunk of the conv
// kernel is one (kh,kw) tap over all 32 channels.
//   awf flat index j per batch decomposes as [mb(4)][kc(9)][half(2)][l16(16)][i(16)]
//   element i of a fragment corresponds to chunk-local K = (half<<4)+i
// ---------------------------------------------------------------------------
__global__ void mix_kernel(const float* __restrict__ conv_w,
                           const float* __restrict__ conv_b,
                           const float* __restrict__ weights,
                           __bf16* __restrict__ awf, float* __restrict__ mbias) {
    const int b = blockIdx.x, tid = threadIdx.x;
    __shared__ float w[EE];
    if (tid < EE) w[tid] = weights[b * EE + tid];
    __syncthreads();
    const int NW = COUT * KDIM;  // 18432
    for (int j = tid; j < NW; j += blockDim.x) {
        const int i    = j & 15;
        const int l16  = (j >> 4) & 15;
        const int half = (j >> 8) & 1;
        const int kc   = (j >> 9) % 9;       // tap index r = kh*3+kw
        const int mb   = (j >> 9) / 9;       // M block (wave)
        const int m    = mb * 16 + l16;      // output channel
        const int ci   = (half << 4) + i;  // chunk-local K = ci
        const int k    = ci * 9 + kc;        // original k = ci*9 + kh*3 + kw
        float acc = 0.f;
#pragma unroll
        for (int e = 0; e < EE; ++e) acc += w[e] * conv_w[e * NW + m * KDIM + k];
        awf[(size_t)b * NW + j] = f2bf(acc);
    }
    if (tid < COUT) {
        float acc = 0.f;
#pragma unroll
        for (int e = 0; e < EE; ++e) acc += w[e] * conv_b[e * COUT + tid];
        mbias[b * COUT + tid] = acc;
    }
}

// ---------------------------------------------------------------------------
// Kernel 3: implicit-GEMM conv via V_WMMA_F32_16X16X32_BF16
//   Per batch: D[64][3969] = Wmix[64][288] x patches[288][3969] (+bias)/wsum
//   Block = 128 threads = 4 wave32. Wave w owns output rows 16w..16w+15.
//   N tile = 64 pixels -> 4 accumulators per wave (A fragment reused 4x).
//   B chunks (32K x 64N, bf16, fragment layout) staged through LDS by all
//   128 threads; K chunk kc == filter tap (kh,kw), channels = chunk-local K.
// ---------------------------------------------------------------------------
__global__ void conv_wmma_kernel(const float* __restrict__ x,
                                 const __bf16* __restrict__ awf,
                                 const float* __restrict__ mbias,
                                 const float* __restrict__ wsum,
                                 float* __restrict__ out) {
    __shared__ v16bf ldsB[128];   // [half(2)][n(64)] -> 16 bf16 (i = chunk-local K low bits)

    const int b    = blockIdx.y;
    const int tile = blockIdx.x;       // 64-pixel N tile
    const int t    = threadIdx.x;
    const int lane = t & 31;
    const int wave = t >> 5;           // M block
    const int half = lane >> 4;
    const int l16  = lane & 15;

    // ---- staging role: thread t stages (sh = t>>6 selects ci group, n = t&63 pixel)
    const int sh = t >> 6;
    const int n  = t & 63;
    int pn = tile * 64 + n;
    if (pn > NPIX - 1) pn = NPIX - 1;  // clamp (stores are guarded later)
    const int ohn = pn / WO;
    const int own = pn - ohn * WO;
    const float* __restrict__ xb = x + (size_t)b * CIN * HW * HW;
    // element i of staged fragment -> ci = sh*16 + i ; per-chunk offset kh*HW+kw
    const float* __restrict__ xs = xb + ((size_t)sh * 16) * HW * HW + (2 * ohn) * HW + 2 * own;

    __builtin_prefetch(xs, 0, 0);

    const __bf16* __restrict__ abase =
        awf + (((size_t)(b * 4 + wave) * 9) * 2 + half) * 16 * 16 + (size_t)l16 * 16;

    v8f acc0 = {}, acc1 = {}, acc2 = {}, acc3 = {};

#pragma unroll
    for (int kc = 0; kc < 9; ++kc) {
        const int kh = kc / 3, kw = kc - 3 * (kc / 3);   // compile-time after unroll
        // gather + convert this thread's 16 channel values for tap (kh,kw)
        v16bf stage;
#pragma unroll
        for (int i = 0; i < 16; ++i) {
            stage[i] = f2bf(xs[(size_t)i * HW * HW + kh * HW + kw]);
        }
        __syncthreads();               // previous chunk's reads complete
        ldsB[t] = stage;
        __syncthreads();               // chunk staged

        // A fragment: one aligned 32B load (pre-packed bf16, fragment layout)
        v16bf afr = *(const v16bf*)(abase + (size_t)kc * 2 * 16 * 16);
        // B fragments for the wave's 4 N sub-tiles
        v16bf b0 = ldsB[half * 64 +  0 + l16];
        v16bf b1 = ldsB[half * 64 + 16 + l16];
        v16bf b2 = ldsB[half * 64 + 32 + l16];
        v16bf b3 = ldsB[half * 64 + 48 + l16];

        acc0 = __builtin_amdgcn_wmma_f32_16x16x32_bf16(false, afr, false, b0, (short)0, acc0, false, false);
        acc1 = __builtin_amdgcn_wmma_f32_16x16x32_bf16(false, afr, false, b1, (short)0, acc1, false, false);
        acc2 = __builtin_amdgcn_wmma_f32_16x16x32_bf16(false, afr, false, b2, (short)0, acc2, false, false);
        acc3 = __builtin_amdgcn_wmma_f32_16x16x32_bf16(false, afr, false, b3, (short)0, acc3, false, false);
    }

    // ---- epilogue: D element r -> M = wave*16 + 8*half + r, N = l16
    const float inv = 1.0f / wsum[b];
    float mbv[8];
#pragma unroll
    for (int r = 0; r < 8; ++r) mbv[r] = mbias[b * COUT + wave * 16 + 8 * half + r];

    const v8f accs[4] = {acc0, acc1, acc2, acc3};
#pragma unroll
    for (int sub = 0; sub < 4; ++sub) {
        const int p = tile * 64 + sub * 16 + l16;
        if (p < NPIX) {
#pragma unroll
            for (int r = 0; r < 8; ++r) {
                const int mo = wave * 16 + 8 * half + r;
                out[((size_t)b * COUT + mo) * NPIX + p] = (accs[sub][r] + mbv[r]) * inv;
            }
        }
    }
}

// ---------------------------------------------------------------------------
extern "C" void kernel_launch(void* const* d_in, const int* in_sizes, int n_in,
                              void* d_out, int out_size, void* d_ws, size_t ws_size,
                              hipStream_t stream) {
    const float* x   = (const float*)d_in[0];
    const float* rv  = (const float*)d_in[1];
    // d_in[2] = task (unused)
    const float* W1  = (const float*)d_in[3];
    const float* b1  = (const float*)d_in[4];
    const float* W2  = (const float*)d_in[5];
    const float* b2  = (const float*)d_in[6];
    const float* cw  = (const float*)d_in[7];
    const float* cb  = (const float*)d_in[8];
    const float* emb = (const float*)d_in[9];
    float* out = (float*)d_out;

    // workspace: weights f32[320] | wsum f32[32] | mbias f32[2048] | awf bf16[32*18432]
    float*  ws      = (float*)d_ws;
    float*  weights = ws;
    float*  wsum    = ws + 320;
    float*  mbias   = ws + 352;
    __bf16* awf     = (__bf16*)((char*)d_ws + 9600);   // 32B-aligned

    routing_kernel<<<BN, 128, 0, stream>>>(rv, W1, b1, W2, b2, emb, weights, wsum);
    mix_kernel<<<BN, 256, 0, stream>>>(cw, cb, weights, awf, mbias);
    conv_wmma_kernel<<<dim3(NT64, BN), 128, 0, stream>>>(x, awf, mbias, wsum, out);
}
